// WindowAttention_9165460209911
// MI455X (gfx1250) — matrix operand order repaired and checked by reading, not compiled, not run on hardware
//
#include <hip/hip_runtime.h>

#define N_TOK 49
#define HD    32
#define NH    16
#define C_DIM 512
#define SCALE 0.1767766952966369f   // 1/sqrt(32)

// ---- LDS layout (bytes) ----
#define Q_OFF   0            // 16 heads * 49 * 32 bf16, stride 64B
#define K_OFF   50176        // same
#define VT_OFF  100352       // 16 heads * 32 * 64 bf16 (d-major, token minor), stride 128B
#define X_OFF   165888       // 64 rows * 512 bf16, row stride 1040B (bank-staggered)  [phase 0/1]
#define P_OFF   165888       // 8 waves * 64*64 bf16 (8192B each)                       [phase 2]
#define O_OFF   231424       // 64 rows * 512 bf16, row stride 1040B
#define SM_BYTES 297984

typedef __attribute__((ext_vector_type(16))) __bf16 v16bf;
typedef __attribute__((ext_vector_type(8)))  float  v8f;

union Frag { v16bf v; uint4 q[2]; };

__device__ __forceinline__ unsigned short f2bf(float f) {
  unsigned u = __float_as_uint(f);
  u += 0x7FFFu + ((u >> 16) & 1u);        // round-to-nearest-even
  return (unsigned short)(u >> 16);
}
__device__ __forceinline__ unsigned pk2(float a, float b) {
  return (unsigned)f2bf(a) | ((unsigned)f2bf(b) << 16);
}

// A-operand fragment: 16x32 bf16 tile from row-major LDS.
// lane L: row = row; VGPR0..7 hold K = hf*16+0..15.
__device__ __forceinline__ v16bf ldA(const unsigned char* base, int strideB,
                                     int row, int kByte, int hf) {
  const unsigned char* p = base + row * strideB + kByte + hf * 32;
  Frag f;
  f.q[0] = *(const uint4*)p;
  f.q[1] = *(const uint4*)(p + 16);
  return f.v;
}
// B-operand fragment whose 16 K-values for this lane are 32 contiguous bytes at p.
__device__ __forceinline__ v16bf ldB32(const unsigned char* p) {
  Frag f;
  f.q[0] = *(const uint4*)p;
  f.q[1] = *(const uint4*)(p + 16);
  return f.v;
}
__device__ __forceinline__ v16bf zfrag() {
  Frag f;
  f.q[0] = make_uint4(0u, 0u, 0u, 0u);
  f.q[1] = make_uint4(0u, 0u, 0u, 0u);
  return f.v;
}

#define WMMA_BF16(a, b, c) \
  __builtin_amdgcn_wmma_f32_16x16x32_bf16(false, (a), false, (b), (short)0, (c), false, false)

// ================= weight pre-swizzle =================
// W (K x N) f32 -> bf16 tiles of 32x16 (K x N). Tile t = nt*(K/32) + kt, 1024B each.
// Lane l owns 32B: dword j = pair ( W[kt*32 + (l/16)*16 + 2j][nt*16 + l%16], W[+1][..] ).
__global__ void swizzle_w(const float* __restrict__ qkv_w,
                          const float* __restrict__ proj_w,
                          unsigned* __restrict__ dst) {
  const int QKV_DW = (C_DIM * 3 * C_DIM) / 2;  // 393216 dwords
  const int PRJ_DW = (C_DIM * C_DIM) / 2;      // 131072 dwords
  for (int g = blockIdx.x * blockDim.x + threadIdx.x; g < QKV_DW + PRJ_DW;
       g += gridDim.x * blockDim.x) {
    const float* W;
    unsigned* outp;
    int Ncols, gg;
    if (g < QKV_DW) { W = qkv_w; Ncols = 3 * C_DIM; gg = g; outp = dst; }
    else            { W = proj_w; Ncols = C_DIM; gg = g - QKV_DW; outp = dst + QKV_DW; }
    int t = gg >> 8, r = gg & 255;
    int lane = r >> 3, j = r & 7;
    int kt = t & 15, nt = t >> 4;
    int n = nt * 16 + (lane & 15);
    int k = kt * 32 + (lane >> 4) * 16 + 2 * j;
    outp[gg] = pk2(W[k * Ncols + n], W[(k + 1) * Ncols + n]);
  }
}

// ================= fused window attention =================
__global__ __launch_bounds__(256, 1) void win_attn(
    const float* __restrict__ x, const float* __restrict__ mask,
    const float* __restrict__ qkv_b, const float* __restrict__ proj_b,
    const float* __restrict__ table, const int* __restrict__ ridx,
    const unsigned short* __restrict__ wq, const unsigned short* __restrict__ wpj,
    float* __restrict__ out) {
  __shared__ __align__(16) unsigned char sm[SM_BYTES];
  const int b = blockIdx.x;
  const int tid = threadIdx.x;
  const int lane = tid & 31, wv = tid >> 5;
  const int hf = lane >> 4, l16 = lane & 15;

  // ---- phase 0: x -> bf16 LDS (row stride 1040B), zero padding rows 49..63 ----
  {
    const float4* xg = (const float4*)(x + (size_t)b * (N_TOK * C_DIM));
    for (int idx = tid; idx < (N_TOK * C_DIM) / 4; idx += 256) {
      float4 f = xg[idx];
      int f0 = idx * 4, row = f0 >> 9, c = f0 & 511;
      *(uint2*)(sm + X_OFF + row * 1040 + c * 2) =
          make_uint2(pk2(f.x, f.y), pk2(f.z, f.w));
    }
    unsigned* xz = (unsigned*)(sm + X_OFF);
    for (int j = N_TOK * 260 + tid; j < 64 * 260; j += 256) xz[j] = 0u;
  }
  __syncthreads();

  // ---- phase 1: QKV GEMM (64x1536x512), route q / k / v^T into LDS ----
  for (int nt = wv; nt < 96; nt += 8) {
    v8f acc[4] = {};
    for (int kk = 0; kk < 16; ++kk) {
      const unsigned char* wb = (const unsigned char*)wq + (nt * 16 + kk) * 1024 + lane * 32;
      v16bf bw = ldB32(wb);
#pragma unroll
      for (int i = 0; i < 4; ++i) {
        v16bf a = ldA(sm + X_OFF, 1040, i * 16 + l16, kk * 64, hf);
        acc[i] = WMMA_BF16(a, bw, acc[i]);
      }
    }
    float bias = qkv_b[nt * 16 + l16];
    int col = nt * 16 + l16;
    int sect = col >> 9, hc = col & 511, h = hc >> 5, d = hc & 31;
#pragma unroll
    for (int i = 0; i < 4; ++i) {
      int r0 = i * 16 + hf * 8;
      if (sect == 0) {
#pragma unroll
        for (int r = 0; r < 8; ++r) {
          int row = r0 + r;
          if (row < N_TOK)
            *(unsigned short*)(sm + Q_OFF + ((h * N_TOK + row) * HD + d) * 2) =
                f2bf((acc[i][r] + bias) * SCALE);
        }
      } else if (sect == 1) {
#pragma unroll
        for (int r = 0; r < 8; ++r) {
          int row = r0 + r;
          if (row < N_TOK)
            *(unsigned short*)(sm + K_OFF + ((h * N_TOK + row) * HD + d) * 2) =
                f2bf(acc[i][r] + bias);
        }
      } else {  // V: transposed store, 8 consecutive tokens -> one b128
        uint4 pk;
        pk.x = pk2(acc[i][0] + bias, acc[i][1] + bias);
        pk.y = pk2(acc[i][2] + bias, acc[i][3] + bias);
        pk.z = pk2(acc[i][4] + bias, acc[i][5] + bias);
        pk.w = pk2(acc[i][6] + bias, acc[i][7] + bias);
        *(uint4*)(sm + VT_OFF + ((h * HD + d) * 64 + r0) * 2) = pk;
      }
    }
  }
  __syncthreads();

  // ---- phase 2: per-head attention (2 heads per wave) ----
  const float* maskw = mask + (size_t)(b & 63) * (N_TOK * N_TOK);
  unsigned char* Pb = sm + P_OFF + wv * 8192;
  for (int hh = 0; hh < 2; ++hh) {
    int h = wv + hh * 8;
    const unsigned char* qh = sm + Q_OFF + h * (N_TOK * HD * 2);
    const unsigned char* kh = sm + K_OFF + h * (N_TOK * HD * 2);
    const unsigned char* vh = sm + VT_OFF + h * (HD * 64 * 2);
#pragma unroll 1
    for (int i = 0; i < 4; ++i) {
      int rowA = i * 16 + l16;
      v16bf aq = (rowA < N_TOK) ? ldA(qh, 64, rowA, 0, hf) : zfrag();
      v8f s[4];
#pragma unroll
      for (int j = 0; j < 4; ++j) {
        int tok = j * 16 + l16;
        v16bf bk = (tok < N_TOK) ? ldB32(kh + tok * 64 + hf * 32) : zfrag();
        v8f z = {};
        s[j] = WMMA_BF16(aq, bk, z);
      }
      // softmax over the 64 columns of each of this tile-row's 16 rows
#pragma unroll
      for (int r = 0; r < 8; ++r) {
        int row = i * 16 + hf * 8 + r;
        float e[4];
#pragma unroll
        for (int j = 0; j < 4; ++j) e[j] = s[j][r];
        if (row < N_TOK) {
          const int* ri = ridx + row * N_TOK;
          const float* mr = maskw + row * N_TOK;
#pragma unroll
          for (int j = 0; j < 4; ++j) {
            int c = j * 16 + l16;
            e[j] = (c < N_TOK) ? (e[j] + table[ri[c] * NH + h] + mr[c]) : -1e30f;
          }
        } else {
#pragma unroll
          for (int j = 0; j < 4; ++j) {
            int c = j * 16 + l16;
            e[j] = (c < N_TOK) ? 0.f : -1e30f;
          }
        }
        float mx = fmaxf(fmaxf(e[0], e[1]), fmaxf(e[2], e[3]));
#pragma unroll
        for (int m = 1; m < 16; m <<= 1) mx = fmaxf(mx, __shfl_xor(mx, m, 32));
        float p[4], ssum = 0.f;
#pragma unroll
        for (int j = 0; j < 4; ++j) { p[j] = __expf(e[j] - mx); ssum += p[j]; }
#pragma unroll
        for (int m = 1; m < 16; m <<= 1) ssum += __shfl_xor(ssum, m, 32);
        float inv = 1.f / ssum;
#pragma unroll
        for (int j = 0; j < 4; ++j)
          *(unsigned short*)(Pb + row * 128 + (j * 16 + l16) * 2) = f2bf(p[j] * inv);
      }
      // P @ V for this tile-row (K = 64 tokens, N = 32 dims)
      v8f o0 = {}, o1 = {};
#pragma unroll
      for (int kk = 0; kk < 2; ++kk) {
        v16bf ap = ldA(Pb, 128, i * 16 + l16, kk * 64, hf);
        v16bf b0 = ldB32(vh + l16 * 128 + kk * 64 + hf * 32);
        v16bf b1 = ldB32(vh + (16 + l16) * 128 + kk * 64 + hf * 32);
        o0 = WMMA_BF16(ap, b0, o0);
        o1 = WMMA_BF16(ap, b1, o1);
      }
      int r0 = i * 16 + hf * 8;
#pragma unroll
      for (int r = 0; r < 8; ++r) {
        unsigned char* orow = sm + O_OFF + (r0 + r) * 1040 + (h * HD) * 2;
        *(unsigned short*)(orow + l16 * 2) = f2bf(o0[r]);
        *(unsigned short*)(orow + (16 + l16) * 2) = f2bf(o1[r]);
      }
    }
  }
  __syncthreads();

  // ---- phase 3: output projection (64x512x512) -> global f32 ----
  float* ob = out + (size_t)b * (N_TOK * C_DIM);
  for (int nt = wv; nt < 32; nt += 8) {
    v8f acc[4] = {};
    for (int kk = 0; kk < 16; ++kk) {
      const unsigned char* wb = (const unsigned char*)wpj + (nt * 16 + kk) * 1024 + lane * 32;
      v16bf bw = ldB32(wb);
#pragma unroll
      for (int i = 0; i < 4; ++i) {
        v16bf a = ldA(sm + O_OFF, 1040, i * 16 + l16, kk * 64, hf);
        acc[i] = WMMA_BF16(a, bw, acc[i]);
      }
    }
    float bias = proj_b[nt * 16 + l16];
    int col = nt * 16 + l16;
#pragma unroll
    for (int i = 0; i < 4; ++i) {
#pragma unroll
      for (int r = 0; r < 8; ++r) {
        int row = i * 16 + hf * 8 + r;
        if (row < N_TOK) ob[row * C_DIM + col] = acc[i][r] + bias;
      }
    }
  }
}

extern "C" void kernel_launch(void* const* d_in, const int* in_sizes, int n_in,
                              void* d_out, int out_size, void* d_ws, size_t ws_size,
                              hipStream_t stream) {
  (void)in_sizes; (void)n_in; (void)out_size; (void)ws_size;
  const float* x      = (const float*)d_in[0];
  const float* mask   = (const float*)d_in[1];
  const float* qkv_w  = (const float*)d_in[2];
  const float* qkv_b  = (const float*)d_in[3];
  const float* proj_w = (const float*)d_in[4];
  const float* proj_b = (const float*)d_in[5];
  const float* table  = (const float*)d_in[6];
  const int*   ridx   = (const int*)d_in[7];
  float* out = (float*)d_out;
  unsigned* wsw = (unsigned*)d_ws;  // [0,393216): qkv_w bf16 swizzled; then proj_w

  swizzle_w<<<1024, 256, 0, stream>>>(qkv_w, proj_w, wsw);
  win_attn<<<4096, 256, 0, stream>>>(
      x, mask, qkv_b, proj_b, table, ridx,
      (const unsigned short*)wsw,
      (const unsigned short*)(wsw + (C_DIM * 3 * C_DIM) / 2), out);
}
